// Pre_FR_44495861186996
// MI455X (gfx1250) — matrix operand order repaired and checked by reading, not compiled, not run on hardware
//
#include <hip/hip_runtime.h>

typedef __attribute__((ext_vector_type(16))) __bf16        v16bf;
typedef __attribute__((ext_vector_type(8)))  float         v8f;
typedef __attribute__((ext_vector_type(4)))  unsigned int  v4u;

#define B_    256
#define FD    2048
#define RR    196
#define DV    300
#define DA    312
#define NC    200
#define KTOP  80
#define RPAD  256   // 16 n-tiles -> exactly 4 tiles per wave, branch-free WMMA loop
#define LDK   40    // padded LDS row stride (bf16 elems), 80 B -> 16-B aligned rows

static __device__ __forceinline__ unsigned short f2bf(float x) {
    unsigned int u = __float_as_uint(x);
    u += 0x7FFFu + ((u >> 16) & 1u);           // round-to-nearest-even
    return (unsigned short)(u >> 16);
}

// ---------------- prep: row inv-norms of V (312 x 300) and att (200 x 312) --------
__global__ void k_norms(const float* __restrict__ V, const float* __restrict__ att,
                        float* __restrict__ invV, float* __restrict__ invAtt) {
    int t = threadIdx.x;
    if (t < DA) {
        float ss = 0.f;
        for (int v = 0; v < DV; ++v) { float x = V[t * DV + v]; ss += x * x; }
        invV[t] = 1.0f / fmaxf(sqrtf(ss), 1e-12f);
    } else if (t < DA + NC) {
        int k = t - DA;
        float ss = 0.f;
        for (int i = 0; i < DA; ++i) { float x = att[k * DA + i]; ss += x * x; }
        invAtt[k] = 1.0f / fmaxf(sqrtf(ss), 1e-12f);
    }
}

// ------------- M_bf16[624 x 2048]: rows 0..311 = Vn*W2, rows 312..623 = Vn*W1 -----
__global__ void k_projmat(const float* __restrict__ V, const float* __restrict__ W1,
                          const float* __restrict__ W2, const float* __restrict__ invV,
                          unsigned short* __restrict__ Mb) {
    int id = blockIdx.x * 256 + threadIdx.x;      // 624*2048 threads
    int i  = id >> 11;                            // / 2048
    int f  = id & 2047;
    const float* Wm  = (i < DA) ? W2 : W1;
    int row          = (i < DA) ? i  : i - DA;
    const float* vr  = V + row * DV;
    float acc = 0.f;
    for (int v = 0; v < DV; ++v) acc += vr[v] * Wm[v * FD + f];
    Mb[(size_t)i * FD + f] = f2bf(acc * invV[row]);
}

// ---------------- per-(b,r) inverse L2 norm of Fs over feature dim ----------------
__global__ void k_fsnorm(const float* __restrict__ Fs, float* __restrict__ invFs) {
    int b = blockIdx.x, r = threadIdx.x;
    if (r >= RR) return;
    const float* p = Fs + (size_t)b * FD * RR + r;
    float ss = 0.f;
    for (int f = 0; f < FD; ++f) { float x = p[(size_t)f * RR]; ss += x * x; }
    invFs[b * RR + r] = 1.0f / fmaxf(sqrtf(ss), 1e-12f);
}

// ---------------- fused: dual GEMM (A_pre & G) + top-K masked softmax + S_p -------
__global__ __launch_bounds__(256)
void k_attn(const float* __restrict__ Fs, const unsigned short* __restrict__ Mb,
            const float* __restrict__ invFs, float* __restrict__ Aout,
            float* __restrict__ Sp) {
    __shared__ unsigned short sA[32 * LDK];       // 32 M-rows (16 A + 16 G) x 32 K
    __shared__ unsigned short sB[RPAD * LDK];     // 256 r-rows x 32 K (k-contig)
    __shared__ float sOut[2][16][RPAD];           // [0]=A_pre, [1]=G

    const int tid  = threadIdx.x;
    const int lane = tid & 31, wave = tid >> 5;
    const int b    = blockIdx.y;
    const int i0   = blockIdx.x * 16;

    // zero sA once (rows with i >= DA stay zero; async copies overwrite valid rows)
    for (int z = tid; z < 32 * LDK; z += 256) sA[z] = 0;
    // zero the padded r rows (196..255) of sB once; never rewritten
    for (int z = tid; z < (RPAD - RR) * LDK; z += 256) sB[RR * LDK + z] = 0;

    const float* fsb = Fs + (size_t)b * FD * RR;
    const float invn = (tid < RR) ? invFs[b * RR + tid] : 0.0f;

    const int mt    = wave & 1;                   // 0 -> A-scores (W2), 1 -> G (W1)
    const int ncol0 = wave >> 1;                  // n-tile stripe start (0..3)
    v8f acc[4] = {};

    const int arow = mt * 16 + (lane & 15);

    // per-thread async A-tile copy: thread -> row m = tid>>3, 8-byte segment tid&7
    const int  am     = tid >> 3;                 // 0..31
    const int  aseg   = tid & 7;                  // 8 B each -> 64 B per row
    const int  aii    = i0 + (am & 15);
    const bool avalid = (aii < DA);
    const int  agrow  = (am < 16) ? aii : (DA + aii);
    unsigned long long agaddr =
        (unsigned long long)(const void*)(Mb + (size_t)agrow * FD) + (unsigned)(aseg * 8);
    const unsigned aldsoff =
        (unsigned)(unsigned long long)(const void*)&sA[am * LDK + aseg * 4];

    for (int ks = 0; ks < 64; ++ks) {
        const int f0 = ks * 32;
        __syncthreads();
        // stage A-tile via async global->LDS DMA (pure bf16 copy, ASYNCcnt-tracked)
        if (avalid) {
            asm volatile("global_load_async_to_lds_b64 %0, %1, off"
                         :: "v"(aldsoff), "v"(agaddr) : "memory");
        }
        agaddr += 64;                             // next 32 bf16 of the K dim
        // stage B-tile: normalize Fs on the fly, store [r][k] (k contiguous)
        if (tid < RR) {
            #pragma unroll 4
            for (int kk = 0; kk < 32; ++kk) {
                float v = fsb[(size_t)(f0 + kk) * RR + tid] * invn;
                sB[tid * LDK + kk] = f2bf(v);
            }
            // prefetch next K-slab of Fs into cache while WMMAs run
            if (f0 + 32 < FD)
                __builtin_prefetch(&fsb[(size_t)(f0 + 32 + (tid & 31)) * RR + tid], 0, 0);
        }
        asm volatile("s_wait_asynccnt 0" ::: "memory");
        __syncthreads();
        // WMMA: A frag  K(j) = 16*(j>>3) + 8*(lane>>4) + (j&7)
        //       B frag  K(j) = 16*(j>>3) + 8*(lane>>4) + (j&7)
        union Frag { v16bf v; v4u q[2]; };
        Frag fa, fb[4];
        const v4u* pa = (const v4u*)(&sA[arow * LDK + ((lane >> 4) << 3)]);
        fa.q[0] = pa[0]; fa.q[1] = pa[2];         // +16 elems = +32 B
        #pragma unroll
        for (int t = 0; t < 4; ++t) {             // issue all B-frag LDS loads first
            const int n = ncol0 + 4 * t;          // 0..15
            const v4u* pb = (const v4u*)(&sB[(n * 16 + (lane & 15)) * LDK + ((lane >> 4) << 3)]);
            fb[t].q[0] = pb[0]; fb[t].q[1] = pb[2];
        }
        #pragma unroll
        for (int t = 0; t < 4; ++t) {             // then 4 back-to-back WMMAs
            acc[t] = __builtin_amdgcn_wmma_f32_16x16x32_bf16(
                false, fa.v, false, fb[t].v, (short)0, acc[t], false, false);
        }
    }
    __syncthreads();
    // dump accumulators: C/D layout row = p + 8*(lane>>4), col = lane&15
    {
        const int rbase = 8 * (lane >> 4);
        const int col   = lane & 15;
        #pragma unroll
        for (int t = 0; t < 4; ++t) {
            const int n = ncol0 + 4 * t;
            #pragma unroll
            for (int p = 0; p < 8; ++p)
                sOut[mt][rbase + p][n * 16 + col] = acc[t][p];
        }
    }
    __syncthreads();

    // ---- phase 2: per row, exact top-80 mask + softmax(A*mask) + S_p -------------
    for (int rr = wave; rr < 16; rr += 8) {
        const int  gi = i0 + rr;
        const bool rowvalid = (gi < DA);
        float a[7]; unsigned int u[7];
        #pragma unroll
        for (int j = 0; j < 7; ++j) {
            int r = lane + 32 * j;
            bool valid = (r < RR);
            float x = valid ? sOut[0][rr][r] : 0.0f;
            a[j] = x;
            unsigned int ub = __float_as_uint(x);
            ub = (ub & 0x80000000u) ? ~ub : (ub | 0x80000000u);   // order-preserving
            u[j] = valid ? ub : 0u;
        }
        // radix-select the 80th-largest ordered value
        unsigned int pre = 0u;
        for (int bit = 31; bit >= 0; --bit) {
            unsigned int test = pre | (1u << bit);
            int cnt = 0;
            #pragma unroll
            for (int j = 0; j < 7; ++j)
                cnt += __popc(__builtin_amdgcn_ballot_w32(u[j] >= test));
            if (cnt >= KTOP) pre = test;
        }
        int cgt = 0; unsigned int eb[7];
        #pragma unroll
        for (int j = 0; j < 7; ++j) {
            cgt  += __popc(__builtin_amdgcn_ballot_w32(u[j] > pre));
            eb[j] = __builtin_amdgcn_ballot_w32(u[j] == pre);
        }
        const int need = KTOP - cgt;
        const unsigned int lmask = (lane == 0) ? 0u : (0xFFFFFFFFu >> (32 - lane));
        int basecnt = 0;
        float x[7];
        #pragma unroll
        for (int j = 0; j < 7; ++j) {
            bool valid = (lane + 32 * j) < RR;
            bool sel = (u[j] > pre) ||
                       ((u[j] == pre) && ((basecnt + __popc(eb[j] & lmask)) < need));
            x[j] = valid ? (sel ? a[j] : 0.0f) : -1e30f;   // A*mask: non-topk -> 0
            basecnt += __popc(eb[j]);
        }
        // softmax over all 196 entries (zeros included, exp(0)=1 terms)
        float mx = -1e30f;
        #pragma unroll
        for (int j = 0; j < 7; ++j) mx = fmaxf(mx, x[j]);
        for (int o = 16; o > 0; o >>= 1) mx = fmaxf(mx, __shfl_xor(mx, o, 32));
        float sm = 0.0f, e[7];
        #pragma unroll
        for (int j = 0; j < 7; ++j) {
            e[j] = __expf(x[j] - mx);
            if (lane + 32 * j >= RR) e[j] = 0.0f;
            sm += e[j];
        }
        for (int o = 16; o > 0; o >>= 1) sm += __shfl_xor(sm, o, 32);
        const float inv = 1.0f / sm;
        float sp = 0.0f;
        #pragma unroll
        for (int j = 0; j < 7; ++j) {
            int r = lane + 32 * j;
            if (r < RR && rowvalid) {
                float av = e[j] * inv;
                Aout[((size_t)b * DA + gi) * RR + r] = av;
                sp += av * sOut[1][rr][r];                 // S_p = Σ_r A_soft * G
            }
        }
        for (int o = 16; o > 0; o >>= 1) sp += __shfl_xor(sp, o, 32);
        if (lane == 0 && rowvalid) Sp[b * DA + gi] = sp;
    }
}

// ---------------- S[b,k] = (Σ_i S_p[b,i] * att[k,i]) * invAtt[k] ------------------
__global__ void k_scores(const float* __restrict__ Sp, const float* __restrict__ att,
                         const float* __restrict__ invAtt, float* __restrict__ Sout) {
    int id = blockIdx.x * 256 + threadIdx.x;
    if (id >= B_ * NC) return;
    int b = id / NC, k = id % NC;
    const float* sp = Sp + b * DA;
    const float* ar = att + k * DA;
    float acc = 0.f;
    for (int i = 0; i < DA; ++i) acc += sp[i] * ar[i];
    Sout[id] = acc * invAtt[k];
}

extern "C" void kernel_launch(void* const* d_in, const int* in_sizes, int n_in,
                              void* d_out, int out_size, void* d_ws, size_t ws_size,
                              hipStream_t stream) {
    (void)in_sizes; (void)n_in; (void)out_size; (void)ws_size;
    const float* Fs  = (const float*)d_in[0];
    const float* V   = (const float*)d_in[1];
    const float* W1  = (const float*)d_in[2];
    const float* W2  = (const float*)d_in[3];
    const float* att = (const float*)d_in[4];

    float* ws      = (float*)d_ws;
    float* invV    = ws;                       // 312
    float* invAtt  = ws + 320;                 // 200
    float* invFs   = ws + 1024;                // 256*196 = 50176
    float* Sp      = ws + 1024 + 50176;        // 256*312 = 79872  (ends @131072)
    unsigned short* Mb = (unsigned short*)(ws + 131072);  // 624*2048 bf16 (~2.6 MB)

    float* Aout = (float*)d_out;                           // 256*312*196
    float* Sout = Aout + (size_t)B_ * DA * RR;             // 256*200

    k_norms  <<<1, 512, 0, stream>>>(V, att, invV, invAtt);
    k_projmat<<<(624 * 2048) / 256, 256, 0, stream>>>(V, W1, W2, invV, Mb);
    k_fsnorm <<<B_, 256, 0, stream>>>(Fs, invFs);
    k_attn   <<<dim3(20, B_), 256, 0, stream>>>(Fs, Mb, invFs, Aout, Sp);
    k_scores <<<(B_ * NC + 255) / 256, 256, 0, stream>>>(Sp, att, invAtt, Sout);
}
